// MyLSTMCell_78855599554847
// MI455X (gfx1250) — matrix operand order repaired and checked by reading, not compiled, not run on hardware
//
#include <hip/hip_runtime.h>
#include <cstdint>

// ---------------------------------------------------------------------------
// MI455X (gfx1250, wave32) fused LSTM cell:
//   pass 1: convert x|h -> bf16 A matrix [B x 2K]   (K = 2048 = D + H)
//   pass 2: convert per-gate weights -> bf16 [4][H x 2K] (row-major n,k)
//   pass 3: fused 4-gate GEMM (v_wmma_f32_16x16x32_bf16) with double-buffered
//           LDS fed by global_load_async_to_lds_b128 (ASYNCcnt pipelining),
//           then bias + tanh + cell-update epilogue -> h_next, c_next.
// ---------------------------------------------------------------------------

#define BATCH   16384
#define HID     1024
#define KDIM    2048            // D + H
#define BM      128             // block tile M
#define BN      64              // block tile N
#define BK      32              // k-step (WMMA K)
#define LDA     40              // padded LDS row stride (bf16 elems)

#if defined(__has_builtin)
#  if __has_builtin(__builtin_amdgcn_global_load_async_to_lds_b128)
#    define HAVE_ASYNC 1
#  endif
#endif
#ifndef HAVE_ASYNC
#  define HAVE_ASYNC 0
#endif

typedef __attribute__((ext_vector_type(16))) __bf16 v16bf;
typedef __attribute__((ext_vector_type(8)))  float  v8f;

union FragBF {
    uint4 u[2];
    v16bf v;
};

#if HAVE_ASYNC
// Builtin signature (from hipcc diagnostic): param 1 is a global-addrspace
// pointer to a 4 x i32 vector; param 2 analogously the LDS destination.
typedef int vi4 __attribute__((ext_vector_type(4)));
typedef __attribute__((address_space(1))) vi4 gvi4;
typedef __attribute__((address_space(3))) vi4 lvi4;

__device__ __forceinline__ void async_b128(const void* g, void* l) {
    // memory -> LDS, 16 bytes per lane, tracked by ASYNCcnt
    __builtin_amdgcn_global_load_async_to_lds_b128(
        (gvi4*)(uintptr_t)g,
        (lvi4*)(unsigned int)(uintptr_t)l,   // flat addr[31:0] == LDS addr
        0, 0);
}
#endif

template <int N>
__device__ __forceinline__ void wait_async() {
#if defined(__has_builtin) && __has_builtin(__builtin_amdgcn_s_wait_asynccnt)
    __builtin_amdgcn_s_wait_asynccnt(N);
#else
    asm volatile("s_wait_asynccnt %0" :: "i"(N) : "memory");
#endif
}

__device__ __forceinline__ unsigned short f2bf(float f) {
    // round-to-nearest-even fp32 -> bf16
    unsigned int u = __float_as_uint(f);
    u += 0x7FFFu + ((u >> 16) & 1u);
    return (unsigned short)(u >> 16);
}

__device__ __forceinline__ uint4 pack8(float4 a, float4 b) {
    uint4 o;
    o.x = (unsigned)f2bf(a.x) | ((unsigned)f2bf(a.y) << 16);
    o.y = (unsigned)f2bf(a.z) | ((unsigned)f2bf(a.w) << 16);
    o.z = (unsigned)f2bf(b.x) | ((unsigned)f2bf(b.y) << 16);
    o.w = (unsigned)f2bf(b.z) | ((unsigned)f2bf(b.w) << 16);
    return o;
}

// ---- pass 1: A = bf16([x | h]), shape [BATCH][KDIM], contiguous ------------
__global__ void lstm_conv_a(const float* __restrict__ x,
                            const float* __restrict__ h,
                            unsigned short* __restrict__ abf) {
    int chunk = blockIdx.x * blockDim.x + threadIdx.x;   // 8 elems per chunk
    int row = chunk >> 8;                                // KDIM/8 = 256 chunks/row
    int c8  = (chunk & 255) * 8;
    const float* src = (c8 < 1024) ? (x + (size_t)row * 1024 + c8)
                                   : (h + (size_t)row * 1024 + (c8 - 1024));
    const float4* s4 = reinterpret_cast<const float4*>(src);
    float4 f0 = s4[0], f1 = s4[1];
    reinterpret_cast<uint4*>(abf)[chunk] = pack8(f0, f1);
}

// ---- pass 2: per-gate weight bf16([Wg_x | Wg_h]), shape [HID][KDIM] --------
__global__ void lstm_conv_w(const float* __restrict__ wx,
                            const float* __restrict__ wh,
                            unsigned short* __restrict__ wbf) {
    int chunk = blockIdx.x * blockDim.x + threadIdx.x;   // 8 elems per chunk
    int n  = chunk >> 8;
    int c8 = (chunk & 255) * 8;
    const float* src = (c8 < 1024) ? (wx + (size_t)n * 1024 + c8)
                                   : (wh + (size_t)n * 1024 + (c8 - 1024));
    const float4* s4 = reinterpret_cast<const float4*>(src);
    float4 f0 = s4[0], f1 = s4[1];
    reinterpret_cast<uint4*>(wbf)[chunk] = pack8(f0, f1);
}

// ---- pass 3: fused 4-gate GEMM + LSTM epilogue -----------------------------
__global__ __launch_bounds__(256)
void lstm_gemm(const unsigned short* __restrict__ abf,   // [BATCH][KDIM] bf16
               const unsigned short* __restrict__ wbf,   // [4][HID][KDIM] bf16
               const float* __restrict__ c_prev,
               const float* __restrict__ bix, const float* __restrict__ bih,
               const float* __restrict__ bfx, const float* __restrict__ bfh,
               const float* __restrict__ bcx, const float* __restrict__ bch,
               const float* __restrict__ box, const float* __restrict__ boh,
               float* __restrict__ out) {
    __shared__ __align__(16) unsigned short As[2][BM][LDA];
    __shared__ __align__(16) unsigned short Bs[2][4][BN][LDA];

    const int tid    = threadIdx.x;
    const int lane   = tid & 31;
    const int wave   = tid >> 5;        // 0..7
    const int waveM  = wave & 3;        // 4 waves along M
    const int waveN  = wave >> 2;       // 2 waves along N
    const int lane16 = lane & 15;
    const int laneHi = lane >> 4;       // 0/1 : selects K-half per ISA layout

    const int bm = blockIdx.x * BM;
    const int bn = blockIdx.y * BN;

    v8f acc[4][2][2] = {};              // [gate][mi][ni] 16x16 f32 frags

    // --- staging maps (invariant across k-steps) ---------------------------
    const int arow = tid >> 1;                // 0..127
    const int ach  = (tid & 1) * 16;          // 0 / 16 (bf16 col)
    const unsigned short* aG = abf + (size_t)(bm + arow) * KDIM + ach;

    int bg[4], bnr[4], bkc[4];
    #pragma unroll
    for (int j = 0; j < 4; ++j) {
        int chunk = tid + j * 256;            // 0..1023, 8 bf16 each
        bg[j]  = chunk >> 8;                  // gate
        int r  = chunk & 255;
        bnr[j] = r >> 2;                      // n within tile
        bkc[j] = (r & 3) * 8;                 // k chunk (bf16)
    }

    auto stage = [&](int buf, int kt) {
#if HAVE_ASYNC
        async_b128(aG + kt,     &As[buf][arow][ach]);
        async_b128(aG + kt + 8, &As[buf][arow][ach + 8]);
        #pragma unroll
        for (int j = 0; j < 4; ++j) {
            const unsigned short* g = wbf + (size_t)bg[j] * (HID * KDIM) +
                                      (size_t)(bn + bnr[j]) * KDIM + kt + bkc[j];
            async_b128(g, &Bs[buf][bg[j]][bnr[j]][bkc[j]]);
        }
#else
        {
            const uint4* g = reinterpret_cast<const uint4*>(aG + kt);
            uint4 d0 = g[0];
            uint4 d1 = g[1];
            uint4* s = reinterpret_cast<uint4*>(&As[buf][arow][ach]);
            s[0] = d0;
            s[1] = d1;
        }
        #pragma unroll
        for (int j = 0; j < 4; ++j) {
            const uint4* src = reinterpret_cast<const uint4*>(
                wbf + (size_t)bg[j] * (HID * KDIM) +
                (size_t)(bn + bnr[j]) * KDIM + kt + bkc[j]);
            *reinterpret_cast<uint4*>(&Bs[buf][bg[j]][bnr[j]][bkc[j]]) = *src;
        }
#endif
    };

    constexpr int NIT = KDIM / BK;      // 64 k-steps
    stage(0, 0);

    #pragma unroll 2
    for (int it = 0; it < NIT; ++it) {
        const int kt  = it * BK;
        const int buf = it & 1;
        if (it + 1 < NIT) stage(buf ^ 1, kt + BK);
#if HAVE_ASYNC
        if (it + 1 < NIT) wait_async<6>();   // current buffer's 6 ops done
        else             wait_async<0>();
#endif
        __syncthreads();                     // all waves' tile data visible

        // A fragments (ISA 16-bit A layout: lane<16 holds K 0..7,16..23 of
        // its row; lane>=16 holds K 8..15,24..31)
        FragBF a[2];
        #pragma unroll
        for (int mi = 0; mi < 2; ++mi) {
            int row = waveM * 32 + mi * 16 + lane16;
            a[mi].u[0] = *reinterpret_cast<const uint4*>(&As[buf][row][8 * laneHi]);
            a[mi].u[1] = *reinterpret_cast<const uint4*>(&As[buf][row][16 + 8 * laneHi]);
        }
        #pragma unroll
        for (int g = 0; g < 4; ++g) {
            #pragma unroll
            for (int ni = 0; ni < 2; ++ni) {
                // B layout: lane = column N, K halves split by laneHi
                int n = waveN * 32 + ni * 16 + lane16;
                FragBF b;
                const uint4* p = reinterpret_cast<const uint4*>(&Bs[buf][g][n][8 * laneHi]);
                b.u[0] = p[0];
                b.u[1] = p[2];
                #pragma unroll
                for (int mi = 0; mi < 2; ++mi) {
                    acc[g][mi][ni] = __builtin_amdgcn_wmma_f32_16x16x32_bf16(
                        false, a[mi].v, false, b.v, (short)0, acc[g][mi][ni],
                        false, false);
                }
            }
        }
        __syncthreads();                     // everyone done reading buf
    }

    // ---- epilogue: bias + tanh + cell update, write h_next / c_next -------
    float* __restrict__ outH = out;
    float* __restrict__ outC = out + (size_t)BATCH * HID;
    const int tileRow0 = bm + waveM * 32;
    const int tileCol0 = bn + waveN * 32;

    #pragma unroll
    for (int ni = 0; ni < 2; ++ni) {
        int col = tileCol0 + ni * 16 + lane16;
        float bI = bix[col] + bih[col];
        float bF = bfx[col] + bfh[col];
        float bC = bcx[col] + bch[col];
        float bO = box[col] + boh[col];
        #pragma unroll
        for (int mi = 0; mi < 2; ++mi) {
            int row0 = tileRow0 + mi * 16 + 8 * laneHi;   // C/D layout: M = vgpr + 8*laneHi
            #pragma unroll
            for (int r = 0; r < 8; ++r) {
                size_t idx = (size_t)(row0 + r) * HID + col;
                float iv = acc[0][mi][ni][r] + bI;
                float fv = acc[1][mi][ni][r] + bF;
                float gv = tanhf(acc[2][mi][ni][r] + bC);
                float ov = acc[3][mi][ni][r] + bO;
                float cn = gv * iv + c_prev[idx] * fv;
                outH[idx] = ov * tanhf(cn);
                outC[idx] = cn;
            }
        }
    }
}

// ---------------------------------------------------------------------------
extern "C" void kernel_launch(void* const* d_in, const int* in_sizes, int n_in,
                              void* d_out, int out_size, void* d_ws, size_t ws_size,
                              hipStream_t stream) {
    const float* x      = (const float*)d_in[0];
    const float* h      = (const float*)d_in[1];
    const float* c_prev = (const float*)d_in[2];

    // setup_inputs order: gates [i,f,c,o] for part x then part h, W then b.
    const float* W[4][2];
    const float* bias[4][2];
    for (int p = 0; p < 2; ++p)
        for (int g = 0; g < 4; ++g) {
            W[g][p]    = (const float*)d_in[3 + 2 * (p * 4 + g)];
            bias[g][p] = (const float*)d_in[4 + 2 * (p * 4 + g)];
        }

    unsigned short* abf = (unsigned short*)d_ws;                      // 64 MB
    unsigned short* wbf = (unsigned short*)((char*)d_ws +
                          (size_t)BATCH * KDIM * sizeof(unsigned short)); // +16 MB

    // pass 1: A = bf16([x|h])  (BATCH*KDIM/8 chunks)
    lstm_conv_a<<<(BATCH * KDIM / 8) / 256, 256, 0, stream>>>(x, h, abf);

    // pass 2: per-gate weights = bf16([Wg_x|Wg_h])
    for (int g = 0; g < 4; ++g)
        lstm_conv_w<<<(HID * KDIM / 8) / 256, 256, 0, stream>>>(
            W[g][0], W[g][1], wbf + (size_t)g * HID * KDIM);

    // pass 3: fused GEMM + LSTM epilogue
    dim3 grid(BATCH / BM, HID / BN);
    lstm_gemm<<<grid, 256, 0, stream>>>(abf, wbf, c_prev,
                                        bias[0][0], bias[0][1],
                                        bias[1][0], bias[1][1],
                                        bias[2][0], bias[2][1],
                                        bias[3][0], bias[3][1],
                                        (float*)d_out);
}
